// MoELayer_8959301779838
// MI455X (gfx1250) — matrix operand order repaired and checked by reading, not compiled, not run on hardware
//
#include <hip/hip_runtime.h>

// MoE FFN (grouped GEMM -> exact GELU -> grouped GEMM) for MI455X / gfx1250.
// Compute-bound (~370 FLOP/byte): cast fp32 -> bf16 on the fly and run on
// v_wmma_f32_16x16x32_bf16. Intermediate activations kept as bf16 in d_ws.
// fp32->bf16x2 packing done with a single v_perm_b32 per pair to keep the
// staging VALU cost far below the WMMA issue rate in the K loop.

typedef __bf16 bf16_t;
typedef __attribute__((ext_vector_type(16))) __bf16 v16bf;
typedef __attribute__((ext_vector_type(8)))  __bf16 v8bf;
typedef __attribute__((ext_vector_type(8)))  float  v8f;
typedef __attribute__((ext_vector_type(4)))  float  v4f;
typedef __attribute__((ext_vector_type(2)))  unsigned v2u;

constexpr int kT = 8192;   // total permuted tokens
constexpr int kH = 2048;   // hidden size
constexpr int kF = 8192;   // ffn hidden size
constexpr int kE = 8;      // experts

constexpr int TM = 128;    // rows per block tile
constexpr int TN = 128;    // cols per block tile
constexpr int TK = 32;     // K depth per stage (one bf16 WMMA K-step)
constexpr int LDSB = 40;   // padded LDS row stride in bf16 (80B: 16B-aligned, bank-skewed)

// Pack two fp32 into one bf16x2 dword: bias for round-half-up, then one
// v_perm_b32 selecting the two high halves. lo lands in bits[15:0].
__device__ __forceinline__ unsigned pack_bf16x2(float lo, float hi) {
  union { float f; unsigned u; } a, b;
  a.f = lo; b.f = hi;
  return __builtin_amdgcn_perm(b.u + 0x8000u, a.u + 0x8000u, 0x07060302u);
}

__device__ __forceinline__ bf16_t f2bf(float f) {
  union { float f; unsigned u; } in; in.f = f;
  union { unsigned short s; bf16_t b; } out;
  out.s = (unsigned short)((in.u + 0x8000u) >> 16);
  return out.b;
}

union FragBF { v16bf v; v8bf h[2]; };

template<bool A_IS_BF16, bool APPLY_GELU>
__global__ __launch_bounds__(256)
void gmm_wmma(const float*  __restrict__ A32,   // [T,K] fp32 (fc1 path)
              const bf16_t* __restrict__ A16,   // [T,K] bf16 (fc2 path)
              const float*  __restrict__ W,     // [E,K,N] fp32
              const int*    __restrict__ tokens_per_expert,
              bf16_t*       __restrict__ out16, // [T,N] bf16 (fc1 path)
              float*        __restrict__ out32, // [T,N] fp32 (fc2 path)
              const int K, const int N)
{
  __shared__ bf16_t sA[2][TM * LDSB];           // row-major  [m][k]
  __shared__ bf16_t sB[2][TN * LDSB];           // transposed [n][k]

  // ---- map blockIdx.x -> (expert, row tile); tiles never straddle experts ----
  int e = -1, row_base = 0, row_end = 0;
  {
    int tacc = 0, racc = 0;
    const int mt = blockIdx.x;
    for (int i = 0; i < kE; ++i) {
      const int c     = tokens_per_expert[i];
      const int tiles = (c + TM - 1) / TM;
      if (e < 0 && mt < tacc + tiles) {
        e = i;
        row_base = racc + (mt - tacc) * TM;
        row_end  = racc + c;
      }
      tacc += tiles; racc += c;
    }
  }
  if (e < 0) return;  // block past last tile (uniform exit)

  const int     n_base = blockIdx.y * TN;
  const float*  We     = W + (size_t)e * (size_t)K * (size_t)N;

  const int tid   = threadIdx.x;
  const int lane  = tid & 31;
  const int wave  = tid >> 5;     // 8 waves
  const int wm    = wave & 1;     // 64-row half of the tile
  const int wn    = wave >> 1;    // 32-col quarter of the tile
  const int lhalf = lane >> 4;    // 0/1
  const int l16   = lane & 15;

  // ---- staging registers (double buffering: global -> regs -> LDS) ----
  v4f  rA32[4];
  v8bf rA16[2];
  v4f  rB0[2], rB1[2];

  auto load_regs = [&](int kt) {
    const int kk = kt * TK;
    if (A_IS_BF16) {
      #pragma unroll
      for (int i = 0; i < 2; ++i) {                 // 512 x v8bf chunks / 256 thr
        const int id = tid + i * 256;
        const int r  = id >> 2;
        const int c  = (id & 3) * 8;
        int grow = row_base + r; if (grow > kT - 1) grow = kT - 1;  // pad rows
        rA16[i] = *(const v8bf*)(A16 + (size_t)grow * K + kk + c);
      }
    } else {
      #pragma unroll
      for (int i = 0; i < 4; ++i) {                 // 1024 x float4 / 256 thr
        const int id = tid + i * 256;
        const int r  = id >> 3;
        const int c  = (id & 7) * 4;
        int grow = row_base + r; if (grow > kT - 1) grow = kT - 1;
        rA32[i] = *(const v4f*)(A32 + (size_t)grow * K + kk + c);
      }
    }
    #pragma unroll
    for (int i = 0; i < 2; ++i) {                   // B: row-pair x 4-col units
      const int u  = tid + i * 256;                 // 0..511
      const int kp = u >> 5;                        // k row pair
      const int cg = u & 31;                        // 4-col group
      const int k  = kp * 2;
      const int n  = cg * 4;
      const float* p = We + (size_t)(kk + k) * N + n_base + n;
      rB0[i] = *(const v4f*)p;
      rB1[i] = *(const v4f*)(p + N);
    }
  };

  auto store_lds = [&](int b) {
    if (A_IS_BF16) {
      #pragma unroll
      for (int i = 0; i < 2; ++i) {
        const int id = tid + i * 256;
        const int r  = id >> 2;
        const int c  = (id & 3) * 8;
        *(v8bf*)&sA[b][r * LDSB + c] = rA16[i];
      }
    } else {
      #pragma unroll
      for (int i = 0; i < 4; ++i) {
        const int id = tid + i * 256;
        const int r  = id >> 3;
        const int c  = (id & 7) * 4;
        v2u p;                                      // 4 bf16 = 2 packed dwords
        p[0] = pack_bf16x2(rA32[i][0], rA32[i][1]);
        p[1] = pack_bf16x2(rA32[i][2], rA32[i][3]);
        *(v2u*)&sA[b][r * LDSB + c] = p;            // ds_store_b64
      }
    }
    #pragma unroll
    for (int i = 0; i < 2; ++i) {                   // transpose B into [n][k]
      const int u  = tid + i * 256;
      const int kp = u >> 5;
      const int cg = u & 31;
      const int k  = kp * 2;
      const int n  = cg * 4;
      #pragma unroll
      for (int c = 0; c < 4; ++c) {                 // pack (k,k+1) -> one dword
        *(unsigned*)&sB[b][(n + c) * LDSB + k] =
            pack_bf16x2(rB0[i][c], rB1[i][c]);      // ds_store_b32
      }
    }
  };

  // ---- accumulators: 4 (M) x 2 (N) fragments of 16x16 f32 ----
  v8f acc[4][2];
  #pragma unroll
  for (int i = 0; i < 4; ++i)
    #pragma unroll
    for (int j = 0; j < 2; ++j)
      #pragma unroll
      for (int r = 0; r < 8; ++r)
        acc[i][j][r] = 0.0f;

  const int nk = K / TK;
  load_regs(0);
  store_lds(0);
  int buf = 0;

  for (int kt = 0; kt < nk; ++kt) {
    __syncthreads();
    const bool more = (kt + 1 < nk);
    if (more) load_regs(kt + 1);    // overlap next tile's global loads w/ WMMA

    // A fragments per ISA layout: row m = lane&15, K chunks at (lane>>4)*8, +16
    FragBF fa[4];
    #pragma unroll
    for (int i = 0; i < 4; ++i) {
      const bf16_t* p = &sA[buf][(wm * 64 + i * 16 + l16) * LDSB + lhalf * 8];
      fa[i].h[0] = *(const v8bf*)p;
      fa[i].h[1] = *(const v8bf*)(p + 16);
    }
    // B fragments: col n = lane&15, K chunks at (lane>>4)*8, +16 (same as A)
    FragBF fb[2];
    #pragma unroll
    for (int j = 0; j < 2; ++j) {
      const bf16_t* p = &sB[buf][(wn * 32 + j * 16 + l16) * LDSB + lhalf * 8];
      fb[j].h[0] = *(const v8bf*)p;
      fb[j].h[1] = *(const v8bf*)(p + 16);
    }

    #pragma unroll
    for (int i = 0; i < 4; ++i)
      #pragma unroll
      for (int j = 0; j < 2; ++j)
        acc[i][j] = __builtin_amdgcn_wmma_f32_16x16x32_bf16(
            false, fa[i].v, false, fb[j].v, (short)0, acc[i][j], false, false);

    if (more) store_lds(buf ^ 1);
    buf ^= 1;
  }

  // ---- epilogue: C layout lane=(N), VGPR r -> M = r + 8*(lane>>4) ----
  #pragma unroll
  for (int i = 0; i < 4; ++i) {
    #pragma unroll
    for (int j = 0; j < 2; ++j) {
      const int col = n_base + wn * 32 + j * 16 + l16;
      #pragma unroll
      for (int r = 0; r < 8; ++r) {
        const int row = row_base + wm * 64 + i * 16 + r + lhalf * 8;
        if (row < row_end) {   // mask padded rows of the last tile per expert
          float x = acc[i][j][r];
          if (APPLY_GELU) {
            x = 0.5f * x * (1.0f + erff(x * 0.70710678118654752f));
            out16[(size_t)row * N + col] = f2bf(x);
          } else {
            out32[(size_t)row * N + col] = x;
          }
        }
      }
    }
  }
}

extern "C" void kernel_launch(void* const* d_in, const int* in_sizes, int n_in,
                              void* d_out, int out_size, void* d_ws, size_t ws_size,
                              hipStream_t stream) {
  const float* x   = (const float*)d_in[0];  // [T,H]
  const float* w1  = (const float*)d_in[1];  // [E,H,F]
  const float* w2  = (const float*)d_in[2];  // [E,F,H]
  const int*   tok = (const int*)d_in[3];    // [E]
  float*   out  = (float*)d_out;             // [T,H]
  bf16_t*  hmid = (bf16_t*)d_ws;             // [T,F] bf16 intermediate (128 MiB)

  dim3 blk(256, 1, 1);
  // grid.x covers worst-case row tiles: T/TM exact tiles + up to E partial tiles
  dim3 g1(kT / TM + kE, kF / TN, 1);  // 72 x 64
  dim3 g2(kT / TM + kE, kH / TN, 1);  // 72 x 16

  // fc1: [T,H] x [E,H,F] -> gelu -> bf16 [T,F]
  gmm_wmma<false, true ><<<g1, blk, 0, stream>>>(x, nullptr, w1, tok, hmid, nullptr, kH, kF);
  // fc2: [T,F] x [E,F,H] -> f32 [T,H]
  gmm_wmma<true,  false><<<g2, blk, 0, stream>>>(nullptr, hmid, w2, tok, nullptr, out, kF, kH);
}
